// PoseRefinerModel_70970039599206
// MI455X (gfx1250) — compile-verified, matrix operand order checked by reading, not run on hardware
//
#include <hip/hip_runtime.h>
#include <hip/hip_bf16.h>

typedef _Float16 half_t;
typedef __attribute__((ext_vector_type(16))) _Float16 v16h;
typedef __attribute__((ext_vector_type(8)))  float    v8f;

#define DEVI static __device__ __forceinline__

// ---------------- problem constants ----------------
constexpr int Nn = 16, Tt = 512, Vv = 33, Cc = 3, Hh = 64;
constexpr int Pp = Nn * Tt * Vv;              // 270336 positions
constexpr float EPSV = 1e-5f;

// ---------------- workspace layout (bytes) ----------------
constexpr size_t BIGBUF   = (size_t)Pp * Hh * 2;          // 34,603,008
constexpr size_t OFF_B0   = 0;
constexpr size_t OFF_B1   = BIGBUF;
constexpr size_t OFF_B2   = 2 * BIGBUF;
constexpr size_t SMALL3   = (size_t)Pp * 3 * 2;           // 1,622,016
constexpr size_t OFF_Y0   = 3 * BIGBUF;
constexpr size_t OFF_G5   = OFF_Y0 + SMALL3;
constexpr size_t OFF_Z5   = OFF_G5 + SMALL3;
constexpr size_t OFF_MGW  = OFF_Z5 + SMALL3;              // 3*64*64 halves
constexpr size_t OFF_TW1  = OFF_MGW + (size_t)3*64*64*2;  // 9*64*64 halves
constexpr size_t OFF_TWM  = OFF_TW1 + (size_t)9*64*64*2;  // 3*9*64*64 halves
constexpr size_t OFF_F32  = OFF_TWM + (size_t)3*9*64*64*2;

// fold-array float offsets (all multiples of 8 -> 32B aligned for float4)
constexpr int FD_S  = 0,    FD_B  = 104;
constexpr int F_B1S1 = 208, F_B1B1 = 272, F_B1S2 = 336, F_B1B2 = 400;
constexpr int F_MS1  = 464, F_MB1  = 656, F_MS2  = 848, F_MB2  = 1040;
constexpr int F_B5S1 = 1232, F_B5B1 = 1240, F_B5S2 = 1248, F_B5B2 = 1256;

struct InPtrs { const float* p[42]; };

// ---------------- helpers ----------------
DEVI v8f zero8() { v8f z = {0.f,0.f,0.f,0.f,0.f,0.f,0.f,0.f}; return z; }

union FragU { uint4 q[2]; v16h h; };
union Pk8   { uint4 q; half_t h[8]; };

DEVI v16h frag16(const half_t* a, const half_t* b) {
    FragU u;
    u.q[0] = *(const uint4*)a;   // halves 0..7  -> K = base+0..7
    u.q[1] = *(const uint4*)b;   // halves 8..15 -> K = base+16..23
    return u.h;
}

DEVI v8f wmma_f16(v16h a, v16h b, v8f c) {
    return __builtin_amdgcn_wmma_f32_16x16x32_f16(
        /*neg_a=*/false, a, /*neg_b=*/false, b,
        /*c_mod=*/(short)0, c, /*reuse_a=*/false, /*reuse_b=*/false);
}

// async Global -> LDS copy of 16 bytes (per active lane); ASYNCcnt-tracked.
// Generic pointers to LDS carry the LDS offset in the low 32 bits (aperture in
// the high dword), so truncation gives the VDST LDS-address operand.
DEVI void async_g2l_b128(void* lds_ptr, const void* gptr) {
    unsigned int  la = (unsigned int)(uintptr_t)lds_ptr;
    unsigned long long ga = (unsigned long long)(uintptr_t)gptr;
    asm volatile("global_load_async_to_lds_b128 %0, %1, off"
                 :: "v"(la), "v"(ga) : "memory");
}

// ---------------- prep: fold BN params, fp16 weight repack ----------------
__global__ void k_prep(InPtrs in, char* ws) {
    float* F = (float*)(ws + OFF_F32);
    int tid = threadIdx.x;
    // data_bn: 99 channels
    for (int i = tid; i < 99; i += 256) {
        float s = in.p[2][i] * rsqrtf(in.p[5][i] + EPSV);
        F[FD_S + i] = s;  F[FD_B + i] = in.p[3][i] - in.p[4][i] * s;
    }
    // block1 BN1/BN2 (+tb fold)
    for (int i = tid; i < 64; i += 256) {
        float s1 = in.p[8][i] * rsqrtf(in.p[11][i] + EPSV);
        F[F_B1S1 + i] = s1;  F[F_B1B1 + i] = in.p[9][i] - in.p[10][i] * s1;
        float s2 = in.p[14][i] * rsqrtf(in.p[17][i] + EPSV);
        F[F_B1S2 + i] = s2;  F[F_B1B2 + i] = (in.p[13][i] - in.p[16][i]) * s2 + in.p[15][i];
    }
    // middle blocks (3,64)
    for (int i = tid; i < 192; i += 256) {
        float s1 = in.p[20][i] * rsqrtf(in.p[23][i] + EPSV);
        F[F_MS1 + i] = s1;  F[F_MB1 + i] = in.p[21][i] - in.p[22][i] * s1;
        float s2 = in.p[26][i] * rsqrtf(in.p[29][i] + EPSV);
        F[F_MS2 + i] = s2;  F[F_MB2 + i] = (in.p[25][i] - in.p[28][i]) * s2 + in.p[27][i];
    }
    // block5 (3,)
    for (int i = tid; i < 3; i += 256) {
        float s1 = in.p[32][i] * rsqrtf(in.p[35][i] + EPSV);
        F[F_B5S1 + i] = s1;  F[F_B5B1 + i] = in.p[33][i] - in.p[34][i] * s1;
        float s2 = in.p[38][i] * rsqrtf(in.p[41][i] + EPSV);
        F[F_B5S2 + i] = s2;  F[F_B5B2 + i] = (in.p[37][i] - in.p[40][i]) * s2 + in.p[39][i];
    }
    // middle graph-conv weights -> fp16 [blk][o][c]
    half_t* MGW = (half_t*)(ws + OFF_MGW);
    for (int i = tid; i < 3*64*64; i += 256) MGW[i] = (half_t)in.p[18][i];
    // b1 temporal weights (o,i,k) -> fp16 [k][o][i]
    half_t* TW1 = (half_t*)(ws + OFF_TW1);
    for (int i = tid; i < 9*64*64; i += 256) {
        int k = i / 4096, o = (i / 64) & 63, c = i & 63;
        TW1[i] = (half_t)in.p[12][(o*64 + c)*9 + k];
    }
    // middle temporal weights (blk,o,i,k) -> fp16 [blk][k][o][i]
    half_t* TWM = (half_t*)(ws + OFF_TWM);
    for (int i = tid; i < 3*9*64*64; i += 256) {
        int blk = i / 36864; int r = i % 36864;
        int k = r / 4096, o = (r / 64) & 63, c = r & 63;
        TWM[i] = (half_t)in.p[24][(size_t)blk*36864 + (o*64 + c)*9 + k];
    }
}

// ---------------- data_bn: x (N,T,V,C) fp32 -> y0 fp16 same layout ----------------
__global__ void __launch_bounds__(256) k_databn(const float* __restrict__ x,
                                                const float* __restrict__ F,
                                                half_t* __restrict__ y0) {
    int idx = blockIdx.x * 256 + threadIdx.x;
    if (idx >= Pp * 3) return;
    int c = idx % 3; int p = idx / 3;
    int v = p % Vv;  int t = (p / Vv) % Tt;
    int q = c * 33 + ((t * 33 + v) >> 9);   // raw .view(N, C*V, T) channel index
    y0[idx] = (half_t)(x[idx] * F[FD_S + q] + F[FD_B + q]);
}

// ---------------- block1 graph conv (C=3 -> H=64), scalar ----------------
__global__ void __launch_bounds__(256) k_gc1(const half_t* __restrict__ y0,
                                             const float* __restrict__ gw,  // (64,3)
                                             const float* __restrict__ gb,
                                             half_t* __restrict__ G) {
    int idx = blockIdx.x * 256 + threadIdx.x;   // Pp*64 exact
    int o = idx & 63; int p = idx >> 6;
    float a = gb[o];
    #pragma unroll
    for (int c = 0; c < 3; c++) a += gw[o*3 + c] * (float)y0[p*3 + c];
    G[idx] = (half_t)a;
}

// ---------------- block5 graph conv (H=64 -> 3), scalar ----------------
__global__ void __launch_bounds__(256) k_gc5(const half_t* __restrict__ Y,
                                             const float* __restrict__ gw,  // (3,64)
                                             const float* __restrict__ gb,
                                             half_t* __restrict__ G5) {
    int idx = blockIdx.x * 256 + threadIdx.x;   // Pp*3 exact
    int o = idx % 3; int p = idx / 3;
    const half_t* yp = Y + (size_t)p * 64;
    float a = gb[o];
    #pragma unroll
    for (int c = 0; c < 64; c++) a += gw[o*64 + c] * (float)yp[c];
    G5[idx] = (half_t)a;
}

// ---------------- adjacency + BN1 + ReLU; G[n,t,v,CH] -> Zr[n,w,t,CH] ----------------
template <int CH>
__global__ void __launch_bounds__(256) k_adj(const half_t* __restrict__ G,
                                             const float* __restrict__ A,
                                             const float* __restrict__ s1,
                                             const float* __restrict__ be1,
                                             half_t* __restrict__ Zr) {
    __shared__ float At[33 * 33];
    __shared__ __align__(16) half_t Gs[8 * 33 * CH];
    int tid = threadIdx.x;
    int n = blockIdx.x >> 6; int t0 = (blockIdx.x & 63) * 8;
    for (int i = tid; i < 1089; i += 256) { int v = i / 33, w = i % 33; At[w*33 + v] = A[i]; }
    const uint4* src = (const uint4*)(G + (size_t)((n * Tt + t0) * Vv) * CH);
    uint4* dst = (uint4*)Gs;
    constexpr int NSEG = 8 * 33 * CH / 8;
    for (int i = tid; i < NSEG; i += 256) dst[i] = src[i];
    __syncthreads();
    for (int i = tid; i < 8 * 33 * CH; i += 256) {
        int tl = i / (33 * CH); int rem = i % (33 * CH);
        int w = rem / CH; int c = rem % CH;
        float acc = 0.f;
        #pragma unroll
        for (int v = 0; v < 33; v++) acc += At[w*33 + v] * (float)Gs[(tl*33 + v)*CH + c];
        float val = acc * s1[c] + be1[c];
        val = val > 0.f ? val : 0.f;
        Zr[((size_t)(n * Vv + w) * Tt + t0 + tl) * CH + c] = (half_t)val;
    }
}

// ---------------- middle graph conv: WMMA GEMM, per-position 64x64 ----------------
__global__ void __launch_bounds__(256) k_gc_wmma(const half_t* __restrict__ Y,
                                                 const half_t* __restrict__ W,   // [o][c] fp16
                                                 const float* __restrict__ gb,
                                                 half_t* __restrict__ G) {
    int lane = threadIdx.x & 31;
    int wid  = (blockIdx.x * 256 + threadIdx.x) >> 5;   // 16896 waves exact
    int ll = lane & 15, hi = lane >> 4;
    int p = wid * 16 + ll;                              // per-lane position (B column)

    const half_t* yb = Y + (size_t)p * 64 + hi * 8;
    v16h b0 = frag16(yb,      yb + 16);                 // K chunk 0 (c 0..31)
    v16h b1 = frag16(yb + 32, yb + 48);                 // K chunk 1 (c 32..63)

    v8f acc[4]; for (int i = 0; i < 4; i++) acc[i] = zero8();
    #pragma unroll
    for (int ot = 0; ot < 4; ot++) {
        const half_t* wb = W + (size_t)((ot*16 + ll) * 64) + hi * 8;
        v16h a0 = frag16(wb,      wb + 16);
        v16h a1 = frag16(wb + 32, wb + 48);
        acc[ot] = wmma_f16(a0, b0, acc[ot]);
        acc[ot] = wmma_f16(a1, b1, acc[ot]);
    }
    #pragma unroll
    for (int ot = 0; ot < 4; ot++) {
        int ob = ot * 16 + hi * 8;
        float4 g0 = *(const float4*)(gb + ob);
        float4 g1 = *(const float4*)(gb + ob + 4);
        float bias[8] = {g0.x,g0.y,g0.z,g0.w,g1.x,g1.y,g1.z,g1.w};
        Pk8 o;
        #pragma unroll
        for (int r = 0; r < 8; r++) o.h[r] = (half_t)(acc[ot][r] + bias[r]);
        *(uint4*)(G + (size_t)p * 64 + ob) = o.q;
    }
}

// ------ temporal conv: 9-tap, H=64 -> H=64, WMMA; + BN2 (+res) + ReLU ------
// Zr layout [n][v][t][c]; Wt layout [k][o][c] fp16; out layout [n][t][v][c]
__global__ void __launch_bounds__(256) k_tconv_wmma(const half_t* __restrict__ Zr,
                                                    const half_t* __restrict__ Wt,
                                                    const float* __restrict__ s2,
                                                    const float* __restrict__ be2,
                                                    const half_t* __restrict__ Yres,
                                                    half_t* __restrict__ Yout,
                                                    int has_res) {
    __shared__ __align__(16) half_t lds[8][24 * 64];    // per-wave 24t x 64c window
    int lane = threadIdx.x & 31;  int wl = threadIdx.x >> 5;
    int wid  = blockIdx.x * 8 + wl;                      // 16896 waves exact
    int tt = wid & 31; int v = (wid >> 5) % 33; int n = wid / (32 * 33);
    int t0 = tt * 16;
    half_t* L = &lds[wl][0];

    // stage input window rows t0-4 .. t0+19 into wave-private LDS:
    // in-range rows via async Global->LDS (no VGPR round-trip, ASYNCcnt),
    // zero-pad rows via direct ds_store.
    const size_t rowbase = (size_t)((n * Vv + v) * Tt);
    #pragma unroll
    for (int i = 0; i < 6; i++) {
        int idx = i * 32 + lane; int row = idx >> 3; int seg = idx & 7;
        int tg = t0 - 4 + row;
        half_t* lp = L + row * 64 + seg * 8;
        if (tg >= 0 && tg < Tt) {
            async_g2l_b128(lp, Zr + (rowbase + tg) * 64 + seg * 8);
        } else {
            uint4 z; z.x = z.y = z.z = z.w = 0u;
            *(uint4*)lp = z;
        }
    }
    asm volatile("s_wait_asynccnt 0\n\t"
                 "s_wait_dscnt 0" ::: "memory");   // LDS visible wave-wide

    int ll = lane & 15, hi = lane >> 4;
    v8f acc[4]; for (int i = 0; i < 4; i++) acc[i] = zero8();

    for (int k = 0; k < 9; k++) {
        const half_t* lb = L + (ll + k) * 64 + hi * 8;   // B column = time, K = channel
        v16h b0 = frag16(lb,      lb + 16);
        v16h b1 = frag16(lb + 32, lb + 48);
        const half_t* wk = Wt + (size_t)k * 4096;
        #pragma unroll
        for (int ot = 0; ot < 4; ot++) {
            const half_t* wb = wk + (ot*16 + ll) * 64 + hi * 8;
            v16h a0 = frag16(wb,      wb + 16);
            v16h a1 = frag16(wb + 32, wb + 48);
            acc[ot] = wmma_f16(a0, b0, acc[ot]);
            acc[ot] = wmma_f16(a1, b1, acc[ot]);
        }
    }

    // epilogue: z*s2 + be2 (tb & BN2 folded) [+res], ReLU, store fp16 [n,t,v,c]
    size_t pos = ((size_t)(n * Tt + t0 + ll) * Vv + v) * 64;
    #pragma unroll
    for (int ot = 0; ot < 4; ot++) {
        int ob = ot * 16 + hi * 8;
        float4 sa = *(const float4*)(s2 + ob),  sbv = *(const float4*)(s2 + ob + 4);
        float4 ba = *(const float4*)(be2 + ob), bb  = *(const float4*)(be2 + ob + 4);
        float sc[8] = {sa.x,sa.y,sa.z,sa.w,sbv.x,sbv.y,sbv.z,sbv.w};
        float bs[8] = {ba.x,ba.y,ba.z,ba.w,bb.x,bb.y,bb.z,bb.w};
        float vals[8];
        #pragma unroll
        for (int r = 0; r < 8; r++) vals[r] = acc[ot][r] * sc[r] + bs[r];
        if (has_res) {
            Pk8 rr; rr.q = *(const uint4*)(Yres + pos + ob);
            #pragma unroll
            for (int r = 0; r < 8; r++) vals[r] += (float)rr.h[r];
        }
        Pk8 o;
        #pragma unroll
        for (int r = 0; r < 8; r++) o.h[r] = (half_t)(vals[r] > 0.f ? vals[r] : 0.f);
        *(uint4*)(Yout + pos + ob) = o.q;
    }
}

// ------ block5 temporal conv (3->3, 9 taps) + BN2 + ReLU + outer residual ------
__global__ void __launch_bounds__(256) k_t5(const half_t* __restrict__ Z5,
                                            const float* __restrict__ tw,   // (3,3,9)
                                            const float* __restrict__ s2,
                                            const float* __restrict__ be2,
                                            const float* __restrict__ x,
                                            float* __restrict__ out) {
    int idx = blockIdx.x * 256 + threadIdx.x;   // Pp exact, over (n,v,t)
    int t = idx & 511; int v = (idx >> 9) % 33; int n = idx / (512 * 33);
    const half_t* zb = Z5 + (size_t)((n * Vv + v) * Tt) * 3;
    float a[3] = {0.f, 0.f, 0.f};
    #pragma unroll
    for (int k = 0; k < 9; k++) {
        int tg = t + k - 4;
        if (tg < 0 || tg >= Tt) continue;
        #pragma unroll
        for (int i = 0; i < 3; i++) {
            float zv = (float)zb[tg*3 + i];
            a[0] += tw[ 0 + i*9 + k] * zv;
            a[1] += tw[27 + i*9 + k] * zv;
            a[2] += tw[54 + i*9 + k] * zv;
        }
    }
    size_t ob = ((size_t)(n * Tt + t) * Vv + v) * 3;
    #pragma unroll
    for (int o = 0; o < 3; o++) {
        float r = a[o] * s2[o] + be2[o];
        r = r > 0.f ? r : 0.f;
        out[ob + o] = r + x[ob + o];
    }
}

// ---------------- host launch ----------------
extern "C" void kernel_launch(void* const* d_in, const int* in_sizes, int n_in,
                              void* d_out, int out_size, void* d_ws, size_t ws_size,
                              hipStream_t stream) {
    (void)in_sizes; (void)n_in; (void)out_size; (void)ws_size;
    char* ws = (char*)d_ws;
    const float* x = (const float*)d_in[0];
    const float* A = (const float*)d_in[1];

    half_t* B0 = (half_t*)(ws + OFF_B0);
    half_t* B1 = (half_t*)(ws + OFF_B1);
    half_t* B2 = (half_t*)(ws + OFF_B2);
    half_t* Y0 = (half_t*)(ws + OFF_Y0);
    half_t* G5 = (half_t*)(ws + OFF_G5);
    half_t* Z5 = (half_t*)(ws + OFF_Z5);
    half_t* MGW = (half_t*)(ws + OFF_MGW);
    half_t* TW1 = (half_t*)(ws + OFF_TW1);
    half_t* TWM = (half_t*)(ws + OFF_TWM);
    float*  F   = (float*)(ws + OFF_F32);

    InPtrs ip;
    for (int i = 0; i < 42; i++) ip.p[i] = (const float*)d_in[i];

    k_prep<<<1, 256, 0, stream>>>(ip, ws);
    k_databn<<<Pp * 3 / 256, 256, 0, stream>>>(x, F, Y0);

    // ---- block 1 (no residual) ----
    k_gc1<<<Pp * 64 / 256, 256, 0, stream>>>(Y0, (const float*)d_in[6], (const float*)d_in[7], B0);
    k_adj<64><<<Nn * (Tt / 8), 256, 0, stream>>>(B0, A, F + F_B1S1, F + F_B1B1, B1);
    k_tconv_wmma<<<2112, 256, 0, stream>>>(B1, TW1, F + F_B1S2, F + F_B1B2, nullptr, B2, 0);

    // ---- middle blocks (residual), buffer rotation ----
    half_t* Yc = B2;
    for (int i = 0; i < 3; i++) {
        half_t* Gm = (i == 0) ? B0 : (i == 1) ? B1 : B2;
        half_t* Zm = (i == 0) ? B1 : (i == 1) ? B2 : B0;
        half_t* Yn = Gm;   // graph-out buffer is free by temporal-conv time
        k_gc_wmma<<<2112, 256, 0, stream>>>(Yc, MGW + (size_t)i * 4096,
                                            (const float*)d_in[19] + i * 64, Gm);
        k_adj<64><<<Nn * (Tt / 8), 256, 0, stream>>>(Gm, A, F + F_MS1 + i * 64,
                                                     F + F_MB1 + i * 64, Zm);
        k_tconv_wmma<<<2112, 256, 0, stream>>>(Zm, TWM + (size_t)i * 36864,
                                               F + F_MS2 + i * 64, F + F_MB2 + i * 64,
                                               Yc, Yn, 1);
        Yc = Yn;
    }

    // ---- block 5 (no residual) + outer residual with x ----
    k_gc5<<<Pp * 3 / 256, 256, 0, stream>>>(Yc, (const float*)d_in[30], (const float*)d_in[31], G5);
    k_adj<3><<<Nn * (Tt / 8), 256, 0, stream>>>(G5, A, F + F_B5S1, F + F_B5B1, Z5);
    k_t5<<<Pp / 256, 256, 0, stream>>>(Z5, (const float*)d_in[36], F + F_B5S2, F + F_B5B2,
                                       x, (float*)d_out);
}
